// SelfAttentionModel_39848706572573
// MI455X (gfx1250) — matrix operand order repaired and checked by reading, not compiled, not run on hardware
//
#include <hip/hip_runtime.h>
#include <hip/hip_bf16.h>

// ---- sizes (compile-time, match reference) ----
#define BATCH 2
#define SEQ 2048
#define EMB 1024
#define NHEAD 16
#define HDIM 64
#define NROW (BATCH * SEQ)          // 4096 rows of x

#define AS1 __attribute__((address_space(1)))
#define AS3 __attribute__((address_space(3)))

typedef __attribute__((ext_vector_type(16))) __bf16 v16bf;
typedef __attribute__((ext_vector_type(8)))  float  v8f;
typedef __attribute__((ext_vector_type(4)))  int    v4i;

union FragC { v8f v; float f[8]; };

__device__ __forceinline__ unsigned short f2bf(float f) {
  union { float f; unsigned u; } x; x.f = f;
  unsigned r = x.u + 0x7FFFu + ((x.u >> 16) & 1u);   // round-to-nearest-even
  return (unsigned short)(r >> 16);
}

// build a 16-element bf16 fragment from two 16-byte chunks
__device__ __forceinline__ v16bf load16(const unsigned short* p0,
                                        const unsigned short* p1) {
  union { v16bf v; uint4 q[2]; } f;
  f.q[0] = *reinterpret_cast<const uint4*>(p0);
  f.q[1] = *reinterpret_cast<const uint4*>(p1);
  return f.v;
}

__device__ __forceinline__ v8f wmma_bf16(v16bf a, v16bf b, v8f c) {
  return __builtin_amdgcn_wmma_f32_16x16x32_bf16(
      /*neg_a=*/false, a, /*neg_b=*/false, b,
      /*c_mod=*/(short)0, c, /*reuse_a=*/false, /*reuse_b=*/false);
}

// ---- CDNA5 async global->LDS copy (16 bytes), tracked with ASYNCcnt ----
__device__ __forceinline__ void async_ld16(unsigned short* lds_p,
                                           const unsigned short* g_p) {
#if __has_builtin(__builtin_amdgcn_global_load_async_to_lds_b128)
  __builtin_amdgcn_global_load_async_to_lds_b128(
      (AS1 v4i*)g_p, (AS3 v4i*)lds_p, 0, 0);
#else
  unsigned loff = (unsigned)(unsigned long long)(AS3 unsigned short*)lds_p;
  unsigned long long ga = (unsigned long long)g_p;
  asm volatile("global_load_async_to_lds_b128 %0, %1, off"
               :: "v"(loff), "v"(ga) : "memory");
#endif
}

__device__ __forceinline__ void wait_async0() {
#if __has_builtin(__builtin_amdgcn_s_wait_asynccnt)
  __builtin_amdgcn_s_wait_asynccnt(0);
#else
  asm volatile("s_wait_asynccnt 0x0" ::: "memory");
#endif
}

// ---------------- kernel 1: cast x (fp32 -> bf16) ----------------
__global__ void cast_x_kernel(const float* __restrict__ src,
                              unsigned short* __restrict__ dst, int n) {
  int i = blockIdx.x * blockDim.x + threadIdx.x;
  if (i < n) dst[i] = f2bf(src[i]);
}

// ------- kernel 2: cast + transpose weights (W[in,out] -> Wt[out,in] bf16) ----
__global__ void cast_wT_kernel(const float* __restrict__ wq,
                               const float* __restrict__ wk,
                               const float* __restrict__ wv,
                               unsigned short* __restrict__ wqt,
                               unsigned short* __restrict__ wkt,
                               unsigned short* __restrict__ wvt) {
  int z = blockIdx.z;
  const float* src = (z == 0) ? wq : (z == 1) ? wk : wv;
  unsigned short* dst = (z == 0) ? wqt : (z == 1) ? wkt : wvt;
  int i = blockIdx.x * blockDim.x + threadIdx.x;   // output index, [out][in]
  if (i < EMB * EMB) {
    int n = i >> 10;           // out column
    int k = i & (EMB - 1);     // in  (K)
    dst[i] = f2bf(src[k * EMB + n]);
  }
}

// ---------------- kernel 3: QKV projection GEMM (WMMA bf16) ----------------
// Each wave: one 16(M) x 64(N) tile. Even/odd register pipeline over K:
// loads land directly in loop-carried regs (no rotate copies).
__global__ __launch_bounds__(128) void qkv_gemm_kernel(
    const unsigned short* __restrict__ xb,   // [4096,1024] bf16
    const unsigned short* __restrict__ wqt,
    const unsigned short* __restrict__ wkt,
    const unsigned short* __restrict__ wvt,  // [1024(out),1024(in)] bf16
    const float* __restrict__ bq, const float* __restrict__ bk,
    const float* __restrict__ bv,
    unsigned short* __restrict__ Qo,         // [b,h,s,d] bf16
    unsigned short* __restrict__ Ko,         // [b,h,s,d] bf16
    unsigned short* __restrict__ Vt)         // [b,h,d,s] bf16
{
  const int z = blockIdx.z;
  const unsigned short* wt = (z == 0) ? wqt : (z == 1) ? wkt : wvt;
  const float* bias        = (z == 0) ? bq  : (z == 1) ? bk  : bv;
  unsigned short* dst      = (z == 0) ? Qo  : (z == 1) ? Ko  : Vt;

  const int lane = threadIdx.x & 31;
  const int wv_  = threadIdx.x >> 5;
  const int mt   = blockIdx.x * 4 + wv_;     // 0..255 row tiles
  const int nb   = blockIdx.y * 64;          // col base
  const int lm   = lane & 15;
  const int g    = lane >> 4;

  const unsigned short* arow = xb + (size_t)(mt * 16 + lm) * EMB + 8 * g;

  auto loadA = [&](int ks) {
    return load16(arow + ks * 32, arow + ks * 32 + 16);
  };
  auto loadB = [&](int nt, int ks) {
    const unsigned short* bp =
        wt + (size_t)(nb + nt * 16 + lm) * EMB + ks * 32 + 8 * g;
    return load16(bp, bp + 16);
  };

  FragC acc[4];
  v8f zero = {};
#pragma unroll
  for (int t = 0; t < 4; ++t) acc[t].v = zero;

  v16bf a0 = loadA(0), a1;
  v16bf b0[4], b1[4];
#pragma unroll
  for (int nt = 0; nt < 4; ++nt) b0[nt] = loadB(nt, 0);

  for (int ks = 0; ks < EMB / 32; ks += 2) {
    a1 = loadA(ks + 1);
#pragma unroll
    for (int nt = 0; nt < 4; ++nt) b1[nt] = loadB(nt, ks + 1);
#pragma unroll
    for (int nt = 0; nt < 4; ++nt)
      acc[nt].v = wmma_bf16(a0, b0[nt], acc[nt].v);

    // final prefetch over-reads <=2KB past the operand; lands in adjacent
    // workspace buffers and is never consumed.
    a0 = loadA(ks + 2);
#pragma unroll
    for (int nt = 0; nt < 4; ++nt) b0[nt] = loadB(nt, ks + 2);
#pragma unroll
    for (int nt = 0; nt < 4; ++nt)
      acc[nt].v = wmma_bf16(a1, b1[nt], acc[nt].v);
  }

#pragma unroll
  for (int nt = 0; nt < 4; ++nt) {
    const int col = nb + nt * 16 + lm;       // = h*64 + d
    const float cb = bias[col];
    const int hh = col >> 6, dd = col & 63;
#pragma unroll
    for (int r = 0; r < 8; ++r) {
      const int row = mt * 16 + r + 8 * g;   // = b*SEQ + s
      const int bb = row >> 11, ss = row & (SEQ - 1);
      const unsigned short v = f2bf(acc[nt].f[r] + cb);
      if (z < 2)
        dst[(((size_t)(bb * NHEAD + hh)) * SEQ + ss) * HDIM + dd] = v;
      else
        dst[(((size_t)(bb * NHEAD + hh)) * HDIM + dd) * SEQ + ss] = v;
    }
  }
}

// ---------------- kernel 4: flash attention ----------------
// 4 waves/block share (b,h): K/V chunks staged once per block into
// double-buffered LDS via async global->LDS copies (ASYNCcnt).
__global__ __launch_bounds__(128) void attn_kernel(
    const unsigned short* __restrict__ Qg,   // [b,h,s,d]
    const unsigned short* __restrict__ Kg,   // [b,h,s,d]
    const unsigned short* __restrict__ Vt,   // [b,h,d,s]
    const int* __restrict__ amask,           // [b,s]
    const int* __restrict__ inv_scale_p,
    float* __restrict__ out)                 // [b,h,s,d] fp32
{
  __shared__ unsigned short Klds[2][32 * HDIM];   // 2 x 4 KB
  __shared__ unsigned short Vlds[2][HDIM * 32];   // 2 x 4 KB
  __shared__ unsigned short Plds[4][16 * 32];     // wave-private P staging

  const int lane = threadIdx.x & 31;
  const int wv_  = threadIdx.x >> 5;
  const int task = blockIdx.x * 4 + wv_;        // 4096 tasks
  const int qt = task & 127;
  const int h  = (task >> 7) & 15;
  const int b  = task >> 11;
  const int lm = lane & 15;
  const int g  = lane >> 4;

  const float scale = 1.0f / (float)inv_scale_p[0];

  const unsigned short* kbase = Kg + ((size_t)(b * NHEAD + h)) * SEQ * HDIM;
  const unsigned short* vbase = Vt + ((size_t)(b * NHEAD + h)) * HDIM * SEQ;
  const int* mrow = amask + b * SEQ;
  unsigned short* pl = Plds[wv_];

  // Q A-fragments for d = 0..31 and 32..63
  const unsigned short* qrow =
      Qg + (((size_t)(b * NHEAD + h)) * SEQ + qt * 16 + lm) * HDIM;
  const v16bf aq0 = load16(qrow + 16 * g,      qrow + 16 * g + 8);
  const v16bf aq1 = load16(qrow + 32 + 16 * g, qrow + 32 + 16 * g + 8);

  float runmax[8], runsum[8];
  FragC o[4];
  v8f zero = {};
#pragma unroll
  for (int r = 0; r < 8; ++r) { runmax[r] = -INFINITY; runsum[r] = 0.0f; }
#pragma unroll
  for (int t = 0; t < 4; ++t) o[t].v = zero;

  // stage one 32-key chunk into LDS buffer bufn (whole block cooperates)
  auto stage = [&](int kc, int bufn) {
    // K chunk: 32 rows x 64 bf16 = 4 KB contiguous
    const unsigned short* kg = kbase + (size_t)kc * 32 * HDIM;
    unsigned short* kl = &Klds[bufn][0];
#pragma unroll
    for (int i = 0; i < 2; ++i) {
      const int off = (threadIdx.x + i * 128) * 8;   // 16B granules
      async_ld16(kl + off, kg + off);
    }
    // V chunk: 64 rows (d) x 32 bf16, row stride SEQ in global
    unsigned short* vl = &Vlds[bufn][0];
#pragma unroll
    for (int i = 0; i < 2; ++i) {
      const int op = threadIdx.x + i * 128;          // 0..255
      const int row = op >> 2, seg = op & 3;
      async_ld16(vl + row * 32 + seg * 8,
                 vbase + (size_t)row * SEQ + kc * 32 + seg * 8);
    }
  };

  stage(0, 0);

  for (int kc = 0; kc < SEQ / 32; ++kc) {
    const int cur = kc & 1;
    wait_async0();
    __syncthreads();                       // buffer `cur` visible to all waves
    if (kc + 1 < SEQ / 32) stage(kc + 1, cur ^ 1);

    // ---- scores from LDS-resident K chunk ----
    FragC s0, s1;
    {
      const unsigned short* kr = &Klds[cur][lm * HDIM];
      v16bf b0 = load16(kr + 16 * g,      kr + 16 * g + 8);
      v16bf b1 = load16(kr + 32 + 16 * g, kr + 32 + 16 * g + 8);
      v8f c = zero;
      c = wmma_bf16(aq0, b0, c);
      c = wmma_bf16(aq1, b1, c);
      s0.v = c;
    }
    {
      const unsigned short* kr = &Klds[cur][(16 + lm) * HDIM];
      v16bf b0 = load16(kr + 16 * g,      kr + 16 * g + 8);
      v16bf b1 = load16(kr + 32 + 16 * g, kr + 32 + 16 * g + 8);
      v8f c = zero;
      c = wmma_bf16(aq0, b0, c);
      c = wmma_bf16(aq1, b1, c);
      s1.v = c;
    }
    const bool ok0 = (mrow[kc * 32 + lm] != 0);
    const bool ok1 = (mrow[kc * 32 + 16 + lm] != 0);
#pragma unroll
    for (int r = 0; r < 8; ++r) {
      s0.f[r] = ok0 ? s0.f[r] * scale : -INFINITY;
      s1.f[r] = ok1 ? s1.f[r] * scale : -INFINITY;
    }

    // ---- online softmax per row (row m = r + 8g; 16-lane half reduce) ----
    float corr[8];
#pragma unroll
    for (int r = 0; r < 8; ++r) {
      float t = fmaxf(s0.f[r], s1.f[r]);
      t = fmaxf(t, __shfl_xor(t, 1, 32));
      t = fmaxf(t, __shfl_xor(t, 2, 32));
      t = fmaxf(t, __shfl_xor(t, 4, 32));
      t = fmaxf(t, __shfl_xor(t, 8, 32));
      const float nm = fmaxf(runmax[r], t);
      corr[r] = __expf(runmax[r] - nm);
      runmax[r] = nm;
      const float p0 = __expf(s0.f[r] - nm);
      const float p1 = __expf(s1.f[r] - nm);
      s0.f[r] = p0; s1.f[r] = p1;
      float rs = p0 + p1;
      rs += __shfl_xor(rs, 1, 32);
      rs += __shfl_xor(rs, 2, 32);
      rs += __shfl_xor(rs, 4, 32);
      rs += __shfl_xor(rs, 8, 32);
      runsum[r] = runsum[r] * corr[r] + rs;
    }

    // rescale O accumulators
#pragma unroll
    for (int nt = 0; nt < 4; ++nt)
#pragma unroll
      for (int r = 0; r < 8; ++r) o[nt].f[r] *= corr[r];

    // stage P (D-layout) to wave-private LDS, reload in A-layout
#pragma unroll
    for (int r = 0; r < 8; ++r) {
      const int m = r + 8 * g;
      pl[m * 32 + lm]      = f2bf(s0.f[r]);
      pl[m * 32 + 16 + lm] = f2bf(s1.f[r]);
    }
    asm volatile("s_wait_dscnt 0" ::: "memory");
    const v16bf pa =
        load16(pl + lm * 32 + 16 * g, pl + lm * 32 + 16 * g + 8);

    // ---- O += P(16x32) @ V(32x64), V from LDS ----
#pragma unroll
    for (int nt = 0; nt < 4; ++nt) {
      const unsigned short* vr = &Vlds[cur][(nt * 16 + lm) * 32 + 16 * g];
      v16bf vb = load16(vr, vr + 8);
      o[nt].v = wmma_bf16(pa, vb, o[nt].v);
    }

    __syncthreads();                 // all waves done with buffer `cur`
  }

  // normalize and store [b,h,s,d] fp32
#pragma unroll
  for (int nt = 0; nt < 4; ++nt)
#pragma unroll
    for (int r = 0; r < 8; ++r) {
      const int q = qt * 16 + r + 8 * g;
      const int d = nt * 16 + lm;
      out[(((size_t)(b * NHEAD + h)) * SEQ + q) * HDIM + d] =
          o[nt].f[r] / runsum[r];
    }
}

// ---------------- host launcher ----------------
extern "C" void kernel_launch(void* const* d_in, const int* in_sizes, int n_in,
                              void* d_out, int out_size, void* d_ws,
                              size_t ws_size, hipStream_t stream) {
  const float* x   = (const float*)d_in[0];
  const int* amask = (const int*)d_in[1];
  const int* invsc = (const int*)d_in[2];
  const float* Wq  = (const float*)d_in[3];
  const float* bq  = (const float*)d_in[4];
  const float* Wk  = (const float*)d_in[5];
  const float* bk  = (const float*)d_in[6];
  const float* Wv  = (const float*)d_in[7];
  const float* bv  = (const float*)d_in[8];
  float* out = (float*)d_out;

  // workspace layout (bf16 buffers), all 256B aligned
  char* ws = (char*)d_ws;
  unsigned short* xb  = (unsigned short*)(ws + 0);          //  8 MB
  unsigned short* wqt = (unsigned short*)(ws + 8388608);    //  2 MB
  unsigned short* wkt = (unsigned short*)(ws + 10485760);   //  2 MB
  unsigned short* wvt = (unsigned short*)(ws + 12582912);   //  2 MB
  unsigned short* Qg  = (unsigned short*)(ws + 14680064);   //  8 MB
  unsigned short* Kg  = (unsigned short*)(ws + 23068672);   //  8 MB
  unsigned short* Vt  = (unsigned short*)(ws + 31457280);   //  8 MB
  (void)in_sizes; (void)n_in; (void)out_size; (void)ws_size;

  const int nx = NROW * EMB;  // 4194304
  cast_x_kernel<<<(nx + 255) / 256, 256, 0, stream>>>(x, xb, nx);

  cast_wT_kernel<<<dim3((EMB * EMB + 255) / 256, 1, 3), 256, 0, stream>>>(
      Wq, Wk, Wv, wqt, wkt, wvt);

  // 256 row-tiles / 4 waves per block, 16 col-groups, z = {Q,K,V}
  qkv_gemm_kernel<<<dim3(64, 16, 3), 128, 0, stream>>>(
      xb, wqt, wkt, wvt, bq, bk, bv, Qg, Kg, Vt);

  // 4096 (b,h,qtile) tasks, 4 waves per block
  attn_kernel<<<1024, 128, 0, stream>>>(Qg, Kg, Vt, amask, invsc, out);
}
